// QuantizedLinear_73478300500194
// MI455X (gfx1250) — matrix operand order repaired and checked by reading, not compiled, not run on hardware
//
#include <hip/hip_runtime.h>

#define OUT_F 11008
#define IN_F  4096
#define BATCH 8

typedef __attribute__((ext_vector_type(16))) _Float16 v16h;
typedef __attribute__((ext_vector_type(8)))  _Float16 v8h;
typedef __attribute__((ext_vector_type(8)))  float    v8f;
typedef __attribute__((ext_vector_type(4)))  float    v4f;
typedef __attribute__((ext_vector_type(4)))  int      v4i;

// ---------- prep: split x into f16 hi/lo planes, compute per-row sums ----------
__global__ __launch_bounds__(256) void x_prep_kernel(
    const float* __restrict__ x, _Float16* __restrict__ xhi,
    _Float16* __restrict__ xlo, float* __restrict__ rowsum)
{
    __shared__ float red[256];
    const int b = blockIdx.x;          // batch row
    const int t = threadIdx.x;
    const float* row = x + (size_t)b * IN_F;
    float s = 0.0f;
    for (int k = t; k < IN_F; k += 256) {
        float v = row[k];
        _Float16 h = (_Float16)v;
        _Float16 l = (_Float16)(v - (float)h);
        xhi[(size_t)b * IN_F + k] = h;
        xlo[(size_t)b * IN_F + k] = l;
        s += (float)h + (float)l;      // sum of the representable x~ = hi+lo
    }
    red[t] = s;
    __syncthreads();
    for (int off = 128; off > 0; off >>= 1) {
        if (t < off) red[t] += red[t + off];
        __syncthreads();
    }
    if (t == 0) rowsum[b] = red[0];
}

// ---------- main: one wave per 16-channel tile, gridDim.y-way K split ----------
// Computes raw  acc[b][col] = sum_k x~[b,k] * q[col,k]  (zp folded out, applied in reduce)
__global__ __launch_bounds__(32) void qlinear_wmma_pre_kernel(
    const _Float16* __restrict__ xhi, const _Float16* __restrict__ xlo,
    const int* __restrict__ q, float* __restrict__ part)
{
    const int lane = threadIdx.x & 31;   // wave32
    const int hsel = lane >> 4;
    const int l16  = lane & 15;
    const int col  = blockIdx.x * 16 + l16;

    const int kchunk = IN_F / (int)gridDim.y;
    const int kbeg   = (int)blockIdx.y * kchunk;
    const int kend   = kbeg + kchunk;

    const int* qrow = q + (size_t)col * IN_F + 8 * hsel;
    // A rows 8-15 are padding; their products only reach C rows we never store.
    const _Float16* hrow = xhi + (size_t)(l16 & (BATCH - 1)) * IN_F + 8 * hsel;
    const _Float16* lrow = xlo + (size_t)(l16 & (BATCH - 1)) * IN_F + 8 * hsel;

    v8f acc0 = {};   // hi-part accumulator
    v8f acc1 = {};   // lo-part accumulator (independent WMMA chains)

    #pragma unroll 2
    for (int k0 = kbeg; k0 < kend; k0 += 32) {
        __builtin_prefetch(qrow + k0 + 32, 0, 1);

        // B tile: raw q -> f16 (exact, |q| <= 128); elem s<8 -> K = k0 + 8*hsel + s, s>=8 -> K = k0 + 16 + 8*hsel + (s-8)
        v4i q0 = *(const v4i*)(qrow + k0);
        v4i q1 = *(const v4i*)(qrow + k0 + 4);
        v4i q2 = *(const v4i*)(qrow + k0 + 16);
        v4i q3 = *(const v4i*)(qrow + k0 + 20);
        v16h bm;
        #pragma unroll
        for (int j = 0; j < 4; ++j) {
            bm[j]      = (_Float16)(short)q0[j];
            bm[4 + j]  = (_Float16)(short)q1[j];
            bm[8 + j]  = (_Float16)(short)q2[j];
            bm[12 + j] = (_Float16)(short)q3[j];
        }

        // A tiles: preconverted packed halves, two contiguous 8-half runs each
        v8h ah0 = *(const v8h*)(hrow + k0);
        v8h ah1 = *(const v8h*)(hrow + k0 + 16);
        v8h al0 = *(const v8h*)(lrow + k0);
        v8h al1 = *(const v8h*)(lrow + k0 + 16);
        v16h ah = __builtin_shufflevector(ah0, ah1, 0,1,2,3,4,5,6,7,8,9,10,11,12,13,14,15);
        v16h al = __builtin_shufflevector(al0, al1, 0,1,2,3,4,5,6,7,8,9,10,11,12,13,14,15);

        acc0 = __builtin_amdgcn_wmma_f32_16x16x32_f16(false, ah, false, bm, (short)0, acc0, false, false);
        acc1 = __builtin_amdgcn_wmma_f32_16x16x32_f16(false, al, false, bm, (short)0, acc1, false, false);
    }

    // C layout: lanes 0-15 hold N=lane, VGPR r -> M=r (the 8 valid batch rows)
    if (hsel == 0) {
        float* p = part + (size_t)blockIdx.y * (BATCH * OUT_F);
        #pragma unroll
        for (int r = 0; r < BATCH; ++r)
            p[(size_t)r * OUT_F + col] = acc0[r] + acc1[r];
    }
}

// ---------- reduce: sum K-splits, apply zero-point correction, scale, bias ----------
__global__ void qlinear_reduce_kernel(const float* __restrict__ part,
                                      const int* __restrict__ zp,
                                      const float* __restrict__ scale,
                                      const float* __restrict__ bias,
                                      const float* __restrict__ rowsum,
                                      float* __restrict__ out, int nsplit)
{
    int t = blockIdx.x * blockDim.x + threadIdx.x;
    if (t >= BATCH * OUT_F) return;
    int o = t % OUT_F;
    int b = t / OUT_F;
    float acc = 0.0f;
    for (int s = 0; s < nsplit; ++s)
        acc += part[(size_t)s * (BATCH * OUT_F) + t];
    acc = fmaf(-(float)zp[o], rowsum[b], acc);   // sum x*(q-zp) = sum x*q - zp*sum x
    out[t] = fmaf(scale[o], acc, bias[o]);
}

// ---------- fallback: fully fused single pass (no workspace needed) ----------
__global__ __launch_bounds__(32) void qlinear_wmma_direct_kernel(
    const float* __restrict__ x, const int* __restrict__ q,
    const int* __restrict__ zp, const float* __restrict__ scale,
    const float* __restrict__ bias, float* __restrict__ out)
{
    const int lane = threadIdx.x & 31;
    const int hsel = lane >> 4;
    const int l16  = lane & 15;
    const int col  = blockIdx.x * 16 + l16;

    const int zprow = zp[col];
    const int*   qrow = q + (size_t)col * IN_F + 8 * hsel;
    const float* xrow = x + (size_t)(l16 & (BATCH - 1)) * IN_F + 8 * hsel;

    v8f acc = {};
    for (int k0 = 0; k0 < IN_F; k0 += 32) {
        __builtin_prefetch(qrow + k0 + 32, 0, 1);
        v4i q0 = *(const v4i*)(qrow + k0);
        v4i q1 = *(const v4i*)(qrow + k0 + 4);
        v4i q2 = *(const v4i*)(qrow + k0 + 16);
        v4i q3 = *(const v4i*)(qrow + k0 + 20);
        v16h bm;
        #pragma unroll
        for (int j = 0; j < 4; ++j) {
            bm[j]      = (_Float16)(short)(q0[j] - zprow);
            bm[4 + j]  = (_Float16)(short)(q1[j] - zprow);
            bm[8 + j]  = (_Float16)(short)(q2[j] - zprow);
            bm[12 + j] = (_Float16)(short)(q3[j] - zprow);
        }
        v4f x0 = *(const v4f*)(xrow + k0);
        v4f x1 = *(const v4f*)(xrow + k0 + 4);
        v4f x2 = *(const v4f*)(xrow + k0 + 16);
        v4f x3 = *(const v4f*)(xrow + k0 + 20);
        v16h ah, al;
        #pragma unroll
        for (int j = 0; j < 4; ++j) {
            { float v = x0[j]; _Float16 h = (_Float16)v; ah[j]      = h; al[j]      = (_Float16)(v - (float)h); }
            { float v = x1[j]; _Float16 h = (_Float16)v; ah[4 + j]  = h; al[4 + j]  = (_Float16)(v - (float)h); }
            { float v = x2[j]; _Float16 h = (_Float16)v; ah[8 + j]  = h; al[8 + j]  = (_Float16)(v - (float)h); }
            { float v = x3[j]; _Float16 h = (_Float16)v; ah[12 + j] = h; al[12 + j] = (_Float16)(v - (float)h); }
        }
        acc = __builtin_amdgcn_wmma_f32_16x16x32_f16(false, ah, false, bm, (short)0, acc, false, false);
        acc = __builtin_amdgcn_wmma_f32_16x16x32_f16(false, al, false, bm, (short)0, acc, false, false);
    }
    if (hsel == 0) {
        const float s = scale[col], bv = bias[col];
        #pragma unroll
        for (int r = 0; r < BATCH; ++r)
            out[(size_t)r * OUT_F + col] = fmaf(s, acc[r], bv);
    }
}

extern "C" void kernel_launch(void* const* d_in, const int* in_sizes, int n_in,
                              void* d_out, int out_size, void* d_ws, size_t ws_size,
                              hipStream_t stream) {
    const float* x     = (const float*)d_in[0];
    const int*   q     = (const int*)  d_in[1];
    const int*   zp    = (const int*)  d_in[2];
    const float* scale = (const float*)d_in[3];
    const float* bias  = (const float*)d_in[4];
    float*       out   = (float*)d_out;

    const int ntiles = OUT_F / 16;   // 688
    const int ksplit = 4;            // 2752 waves -> enough to saturate HBM

    const size_t partBytes = (size_t)ksplit * BATCH * OUT_F * sizeof(float);
    const size_t planeBytes = (size_t)BATCH * IN_F * sizeof(_Float16);
    const size_t need = partBytes + 2 * planeBytes + BATCH * sizeof(float);

    if (ws_size >= need) {
        char* w = (char*)d_ws;
        float*    part   = (float*)w;             w += partBytes;
        _Float16* xhi    = (_Float16*)w;          w += planeBytes;
        _Float16* xlo    = (_Float16*)w;          w += planeBytes;
        float*    rowsum = (float*)w;

        x_prep_kernel<<<BATCH, 256, 0, stream>>>(x, xhi, xlo, rowsum);

        dim3 grid(ntiles, ksplit);
        qlinear_wmma_pre_kernel<<<grid, 32, 0, stream>>>(xhi, xlo, q, part);

        const int n = BATCH * OUT_F;
        qlinear_reduce_kernel<<<(n + 255) / 256, 256, 0, stream>>>(
            part, zp, scale, bias, rowsum, out, ksplit);
    } else {
        qlinear_wmma_direct_kernel<<<ntiles, 32, 0, stream>>>(x, q, zp, scale, bias, out);
    }
}
